// RationaleSelectorModel_13460427505851
// MI455X (gfx1250) — matrix operand order repaired and checked by reading, not compiled, not run on hardware
//
#include <hip/hip_runtime.h>
#include <hip/hip_bf16.h>

// ---------------------------------------------------------------------------
// RationaleSelectorModel fused pipeline for gfx1250 (MI455X, wave32, WMMA).
// Shapes: B=32, T=1024, D=256, H=341 (pad 352), R=4, V=32000.
// HBM-bound (~100MB traffic @23.3TB/s); GEMM done with split-bf16 (3x wmma)
// v_wmma_f32_16x16x32_bf16 for ~f32 accuracy at negligible extra cost.
// ---------------------------------------------------------------------------

#define B_  32
#define T_  1024
#define D_  256
#define H_  341
#define HP_ 352          // H padded to multiple of 16
#define R_  4

typedef __bf16 bf16_t;
typedef bf16_t v16bf __attribute__((ext_vector_type(16)));
typedef float  v8f   __attribute__((ext_vector_type(8)));

union Frag { v16bf v; unsigned u[8]; };

__device__ __forceinline__ unsigned short f32_bf16(float f) {
  unsigned x = __float_as_uint(f);
  unsigned r = (x + 0x7FFFu + ((x >> 16) & 1u)) >> 16;   // round-to-nearest-even
  return (unsigned short)r;
}
__device__ __forceinline__ float bf16_f32(unsigned short h) {
  return __uint_as_float(((unsigned)h) << 16);
}
__device__ __forceinline__ unsigned packpair(float a, float b, float* ra, float* rb) {
  unsigned short ah = f32_bf16(a), bh = f32_bf16(b);
  *ra = a - bf16_f32(ah);
  *rb = b - bf16_f32(bh);
  return (unsigned)ah | ((unsigned)bh << 16);
}

// ---------------------------------------------------------------------------
// K0: split w1 (f32, [256][341]) into pair-packed bf16 hi/lo [128][352];
// pad b1/w2 to 352 with zeros (padded cols then give gelu(0)*0 = 0).
// ---------------------------------------------------------------------------
__global__ void k_prep(const float* __restrict__ w1, const float* __restrict__ b1,
                       const float* __restrict__ w2,
                       unsigned* __restrict__ w1hi, unsigned* __restrict__ w1lo,
                       float* __restrict__ b1p, float* __restrict__ w2p) {
  int idx = blockIdx.x * blockDim.x + threadIdx.x;
  if (idx < HP_) {
    b1p[idx] = (idx < H_) ? b1[idx] : 0.f;
    w2p[idx] = (idx < H_) ? w2[idx] : 0.f;
  }
  if (idx >= 128 * HP_) return;
  int kk = idx / HP_, n = idx % HP_;
  float a = (n < H_) ? w1[(2 * kk) * H_ + n]     : 0.f;
  float b = (n < H_) ? w1[(2 * kk + 1) * H_ + n] : 0.f;
  float ra, rb;
  unsigned hi = packpair(a, b, &ra, &rb);
  float d0, d1;
  unsigned lo = packpair(ra, rb, &d0, &d1);
  w1hi[idx] = hi;
  w1lo[idx] = lo;
}

// ---------------------------------------------------------------------------
// K1: fused  LN(emb*sel) -> split-bf16 WMMA GEMM (x w1) -> +b1 -> exact GELU
//     -> dot w2 -> +b2 -> mask  => scores[B*T].
// One wave per 16-token tile; 2 waves per block. A fragments built once in
// the ISA 16-bit A(16x32) layout from LDS; B tiles loaded pair-packed.
// ---------------------------------------------------------------------------
__global__ __launch_bounds__(64) void k_mlp(
    const float* __restrict__ emb, const float* __restrict__ attn,
    const float* __restrict__ lnw, const float* __restrict__ lnb,
    const unsigned* __restrict__ w1hi, const unsigned* __restrict__ w1lo,
    const float* __restrict__ b1p, const float* __restrict__ w2p,
    const float* __restrict__ b2, float* __restrict__ scores) {
  __shared__ unsigned sHi[2][16][132];   // padded rows: stride 132 dwords -> no 64-bank conflict
  __shared__ unsigned sLo[2][16][132];
  const int wave = threadIdx.x >> 5;
  const int lane = threadIdx.x & 31;
  const int tile = blockIdx.x * 2 + wave;
  const int tok0 = tile * 16;
  const int l8 = lane * 8;

  float wv[8], bv[8];
#pragma unroll
  for (int j = 0; j < 8; j++) { wv[j] = lnw[l8 + j]; bv[j] = lnb[l8 + j]; }

  // ---- Phase 1: per-token layernorm, split to bf16 hi/lo pairs in LDS ----
  for (int m = 0; m < 16; m++) {
    const int tok = tok0 + m;
    const float sel = attn[tok];
    const float* row = emb + (size_t)tok * D_ + l8;
    float x[8]; float s = 0.f, ss = 0.f;
#pragma unroll
    for (int j = 0; j < 8; j++) { float v = row[j] * sel; x[j] = v; s += v; ss += v * v; }
#pragma unroll
    for (int o = 16; o >= 1; o >>= 1) { s += __shfl_xor(s, o, 32); ss += __shfl_xor(ss, o, 32); }
    const float mean = s * (1.f / D_);
    const float var  = ss * (1.f / D_) - mean * mean;
    const float rstd = rsqrtf(var + 1e-5f);
#pragma unroll
    for (int jj = 0; jj < 4; jj++) {
      float a = (x[2 * jj]     - mean) * rstd * wv[2 * jj]     + bv[2 * jj];
      float c = (x[2 * jj + 1] - mean) * rstd * wv[2 * jj + 1] + bv[2 * jj + 1];
      float ra, rc;
      sHi[wave][m][lane * 4 + jj] = packpair(a, c, &ra, &rc);
      float d0, d1;
      sLo[wave][m][lane * 4 + jj] = packpair(ra, rc, &d0, &d1);
    }
  }
  __syncthreads();

  // ---- Phase 2: build A fragments (16-bit A 16x32 layout; 8 k-steps) ----
  const int M = lane & 15, half = lane >> 4;
  Frag ahi[8], alo[8];
#pragma unroll
  for (int ks = 0; ks < 8; ks++) {
#pragma unroll
    for (int j = 0; j < 8; j++) {
      // VGPR j holds K pair: j<4 -> K=2j+8*half ; j>=4 -> K=16+2(j-4)+8*half
      const int p = (j < 4) ? (j + 4 * half) : (8 + (j - 4) + 4 * half);
      ahi[ks].u[j] = sHi[wave][M][ks * 16 + p];
      alo[ks].u[j] = sLo[wave][M][ks * 16 + p];
    }
  }

  // ---- Phase 3: N-tile loop (22 tiles of 16 cols), fused GELU + w2 dot ----
  float sacc[8];
#pragma unroll
  for (int v = 0; v < 8; v++) sacc[v] = 0.f;

  for (int nt = 0; nt < 22; nt++) {
    const int n = nt * 16 + (lane & 15);
    const float b1v = b1p[n], w2v = w2p[n];
    v8f c = {};
#pragma unroll
    for (int ks = 0; ks < 8; ks++) {
      Frag bhi, blo;
#pragma unroll
      for (int j = 0; j < 8; j++) {
        // B(32x16): VGPR j holds K pair: j<4 -> K=2j+8*half ; j>=4 -> K=16+2(j-4)+8*half
        const int kk = ks * 16 + ((j < 4) ? (j + 4 * half) : (8 + (j - 4) + 4 * half));
        bhi.u[j] = w1hi[kk * HP_ + n];
        blo.u[j] = w1lo[kk * HP_ + n];
      }
      // split-bf16: hi*hi + hi*lo + lo*hi  (~f32 accuracy)
      c = __builtin_amdgcn_wmma_f32_16x16x32_bf16(false, ahi[ks].v, false, bhi.v, (short)0, c, false, false);
      c = __builtin_amdgcn_wmma_f32_16x16x32_bf16(false, ahi[ks].v, false, blo.v, (short)0, c, false, false);
      c = __builtin_amdgcn_wmma_f32_16x16x32_bf16(false, alo[ks].v, false, bhi.v, (short)0, c, false, false);
    }
#pragma unroll
    for (int v = 0; v < 8; v++) {
      const float xx = c[v] + b1v;                          // C layout: row M=v+8*half, col N=lane%16
      const float g  = 0.5f * xx * (1.f + erff(xx * 0.70710678118654752f));
      sacc[v] += g * w2v;
    }
  }

  // ---- Phase 4: reduce over N (16 lanes per half), add b2, mask, store ----
  const float b2v = b2[0];
#pragma unroll
  for (int v = 0; v < 8; v++) {
    float s = sacc[v];
#pragma unroll
    for (int o = 8; o >= 1; o >>= 1) s += __shfl_xor(s, o, 32);
    if ((lane & 15) == 0) {
      const int tok = tok0 + v + 8 * half;
      const float sc = s + b2v;
      scores[tok] = (attn[tok] != 0.f) ? sc : 0.f;
    }
  }
}

// ---------------------------------------------------------------------------
// K2: full_rep[b][d] = sum_t attn*emb / max(sum attn,1e-8);  T_eff[b]=sum attn
// ---------------------------------------------------------------------------
__global__ __launch_bounds__(256) void k_pool(const float* __restrict__ emb,
                                              const float* __restrict__ attn,
                                              float* __restrict__ full_rep,
                                              float* __restrict__ t_eff) {
  __shared__ float sa[T_];
  __shared__ float red[256];
  const int b = blockIdx.x, tid = threadIdx.x;
  float psum = 0.f;
  for (int t = tid; t < T_; t += 256) { float a = attn[b * T_ + t]; sa[t] = a; psum += a; }
  red[tid] = psum; __syncthreads();
  for (int o = 128; o >= 1; o >>= 1) { if (tid < o) red[tid] += red[tid + o]; __syncthreads(); }
  const float wsum = red[0];
  if (tid == 0) t_eff[b] = wsum;
  const float inv = 1.f / fmaxf(wsum, 1e-8f);
  float acc = 0.f;
  const float* base = emb + (size_t)b * T_ * D_ + tid;   // tid == d (blockDim==D)
  for (int t = 0; t < T_; t++) acc += sa[t] * base[(size_t)t * D_];
  full_rep[b * D_ + tid] = acc * inv;
}

// ---------------------------------------------------------------------------
// K3: weighted z-score of scores, then ranks[j] = 1 + a_j * sum_i sig((z_j-z_i)/tau)^2
// ---------------------------------------------------------------------------
__global__ __launch_bounds__(1024) void k_rank(const float* __restrict__ scores,
                                               const float* __restrict__ attn,
                                               float* __restrict__ ranks) {
  __shared__ float sz[T_];
  __shared__ float red[1024];
  const int b = blockIdx.x, t = threadIdx.x;
  const float a = attn[b * T_ + t];
  const float s = scores[b * T_ + t];          // already masked
  red[t] = a; __syncthreads();
  for (int o = 512; o >= 1; o >>= 1) { if (t < o) red[t] += red[t + o]; __syncthreads(); }
  const float denom = fmaxf(red[0], 1.f);
  __syncthreads();
  red[t] = s * a; __syncthreads();
  for (int o = 512; o >= 1; o >>= 1) { if (t < o) red[t] += red[t + o]; __syncthreads(); }
  const float mean = red[0] / denom;
  __syncthreads();
  const float dm = s - mean;
  red[t] = dm * dm * a; __syncthreads();
  for (int o = 512; o >= 1; o >>= 1) { if (t < o) red[t] += red[t + o]; __syncthreads(); }
  const float var = red[0] / denom;
  __syncthreads();
  const float z = dm * rsqrtf(var + 1e-6f);
  sz[t] = z; __syncthreads();
  float acc = 0.f;
  for (int i = 0; i < T_; i++) {
    const float x = (z - sz[i]) * 20.f;        // 1/tau_rank = 20
    const float sg = 1.f / (1.f + expf(-x));
    acc += sg * sg;                            // gamma = 2
  }
  ranks[b * T_ + t] = (a == 0.f) ? 1e9f : (1.f + a * acc);
}

// ---------------------------------------------------------------------------
// K5: counting-rank (== stable argsort position), hard top-k, g_soft,
//     g_st = (hard+g_soft)-g_soft (forward value of straight-through), rho_eff.
// ---------------------------------------------------------------------------
__global__ __launch_bounds__(1024) void k_gate(const float* __restrict__ ranks,
                                               const float* __restrict__ attn,
                                               const float* __restrict__ t_eff,
                                               float* __restrict__ gst_all,
                                               float* __restrict__ d_hard,
                                               float* __restrict__ d_gst_last,
                                               float* __restrict__ d_rhoeff) {
  __shared__ float sr[T_];
  __shared__ float red[1024];
  __shared__ float kv[R_];
  const int b = blockIdx.x, t = threadIdx.x;
  const float rk = ranks[b * T_ + t];
  const float a = attn[b * T_ + t];
  sr[t] = rk;
  if (t == 0) {
    const float te = t_eff[b];
    const float rh[R_] = {0.1f, 0.25f, 0.5f, 0.75f};
    for (int r = 0; r < R_; r++) {
      float k = rintf(rh[r] * te);             // ties-to-even, matches jnp.round
      kv[r] = (te > 0.f) ? fmaxf(k, 1.f) : 0.f;
    }
  }
  __syncthreads();
  int pos = 0;
  for (int i = 0; i < T_; i++) {
    const float ri = sr[i];
    pos += (ri < rk) || (ri == rk && i < t);
  }
  const float te1 = fmaxf(t_eff[b], 1.f);
  for (int r = 0; r < R_; r++) {
    const float k = kv[r];
    const float hard = ((float)pos < k) ? 1.f : 0.f;
    const float graw = (1.f / (1.f + expf(-(k - rk) * 5.f))) * a;   // 1/tau_gate = 5
    red[t] = graw; __syncthreads();
    for (int o = 512; o >= 1; o >>= 1) { if (t < o) red[t] += red[t + o]; __syncthreads(); }
    const float gsum = red[0];
    __syncthreads();
    const float gs = graw / fmaxf(gsum, 1e-8f) * k;
    const float gst = (hard + gs) - gs;        // forward of straight-through
    const int oidx = (r * B_ + b) * T_ + t;
    d_hard[oidx] = hard;
    gst_all[oidx] = gst;
    if (r == R_ - 1) d_gst_last[b * T_ + t] = gst;
    red[t] = hard; __syncthreads();
    for (int o = 512; o >= 1; o >>= 1) { if (t < o) red[t] += red[t + o]; __syncthreads(); }
    if (t == 0) d_rhoeff[r * B_ + b] = red[0] / te1;
    __syncthreads();
  }
}

// ---------------------------------------------------------------------------
// K6: pred[r][b][d] = sum_t (attn*g_st) * embed_table[ids] / max(sum w,1e-8)
//     (single pass over the 33.5MB gather for all 4 rhos; prefetch ahead)
// ---------------------------------------------------------------------------
__global__ __launch_bounds__(256) void k_pred(const int* __restrict__ ids,
                                              const float* __restrict__ attn,
                                              const float* __restrict__ gst_all,
                                              const float* __restrict__ table,
                                              float* __restrict__ pred) {
  __shared__ float4 sw[T_];
  __shared__ int    sid[T_];
  __shared__ float  swsum[R_];
  const int b = blockIdx.x, tid = threadIdx.x;
  if (tid < R_) swsum[tid] = 0.f;
  __syncthreads();
  float l0 = 0.f, l1 = 0.f, l2 = 0.f, l3 = 0.f;
  for (int t = tid; t < T_; t += 256) {
    const float a = attn[b * T_ + t];
    float4 w;
    w.x = a * gst_all[(0 * B_ + b) * T_ + t];
    w.y = a * gst_all[(1 * B_ + b) * T_ + t];
    w.z = a * gst_all[(2 * B_ + b) * T_ + t];
    w.w = a * gst_all[(3 * B_ + b) * T_ + t];
    sw[t] = w;
    sid[t] = ids[b * T_ + t];
    l0 += w.x; l1 += w.y; l2 += w.z; l3 += w.w;
  }
  atomicAdd(&swsum[0], l0); atomicAdd(&swsum[1], l1);
  atomicAdd(&swsum[2], l2); atomicAdd(&swsum[3], l3);
  __syncthreads();
  const int d = tid;
  float a0 = 0.f, a1 = 0.f, a2 = 0.f, a3 = 0.f;
  for (int t = 0; t < T_; t++) {
    __builtin_prefetch(&table[(size_t)sid[(t + 6) & (T_ - 1)] * D_ + d], 0, 0);
    const float e = table[(size_t)sid[t] * D_ + d];
    const float4 w = sw[t];
    a0 += w.x * e; a1 += w.y * e; a2 += w.z * e; a3 += w.w * e;
  }
  pred[(0 * B_ + b) * D_ + d] = a0 / fmaxf(swsum[0], 1e-8f);
  pred[(1 * B_ + b) * D_ + d] = a1 / fmaxf(swsum[1], 1e-8f);
  pred[(2 * B_ + b) * D_ + d] = a2 / fmaxf(swsum[2], 1e-8f);
  pred[(3 * B_ + b) * D_ + d] = a3 / fmaxf(swsum[3], 1e-8f);
}

// ---------------------------------------------------------------------------
// K7: per_sample[r][b] = 1 - cos(pred, full_rep)
// ---------------------------------------------------------------------------
__global__ __launch_bounds__(256) void k_cos(const float* __restrict__ pred,
                                             const float* __restrict__ full_rep,
                                             float* __restrict__ per_sample) {
  __shared__ float r0[256], r1[256], r2[256];
  const int rb = blockIdx.x;
  const int b = rb % B_;
  const int d = threadIdx.x;
  const float p = pred[rb * D_ + d];
  const float f = full_rep[b * D_ + d];
  r0[d] = p * f; r1[d] = p * p; r2[d] = f * f;
  __syncthreads();
  for (int o = 128; o >= 1; o >>= 1) {
    if (d < o) { r0[d] += r0[d + o]; r1[d] += r1[d + o]; r2[d] += r2[d + o]; }
    __syncthreads();
  }
  if (d == 0) {
    const float den = fmaxf(sqrtf(r1[0]), 1e-8f) * fmaxf(sqrtf(r2[0]), 1e-8f);
    per_sample[rb] = 1.f - r0[0] / den;
  }
}

// ---------------------------------------------------------------------------
// K8: recon_avg (scalar) and per-rho means
// ---------------------------------------------------------------------------
__global__ void k_final(const float* __restrict__ per_sample,
                        float* __restrict__ out_recon, float* __restrict__ out_perr) {
  __shared__ float red[R_ * B_];
  const int i = threadIdx.x;
  const float v = per_sample[i];
  red[i] = v; __syncthreads();
  for (int o = 64; o >= 1; o >>= 1) { if (i < o) red[i] += red[i + o]; __syncthreads(); }
  if (i == 0) out_recon[0] = red[0] / (float)(R_ * B_);
  if (i < R_) {
    float s = 0.f;
    for (int b = 0; b < B_; b++) s += per_sample[i * B_ + b];
    out_perr[i] = s / (float)B_;
  }
}

// ---------------------------------------------------------------------------
extern "C" void kernel_launch(void* const* d_in, const int* in_sizes, int n_in,
                              void* d_out, int out_size, void* d_ws, size_t ws_size,
                              hipStream_t stream) {
  const int*   ids   = (const int*)  d_in[0];
  const float* emb   = (const float*)d_in[1];
  const float* attn  = (const float*)d_in[2];
  const float* lnw   = (const float*)d_in[3];
  const float* lnb   = (const float*)d_in[4];
  const float* w1    = (const float*)d_in[5];
  const float* b1    = (const float*)d_in[6];
  const float* w2    = (const float*)d_in[7];
  const float* b2    = (const float*)d_in[8];
  const float* table = (const float*)d_in[9];

  char* ws = (char*)d_ws;
  size_t off = 0;
  auto alloc = [&](size_t bytes) { char* p = ws + off; off += (bytes + 255) & ~(size_t)255; return p; };
  unsigned* w1hi   = (unsigned*)alloc(128 * HP_ * 4);
  unsigned* w1lo   = (unsigned*)alloc(128 * HP_ * 4);
  float*    b1p    = (float*)   alloc(HP_ * 4);
  float*    w2p    = (float*)   alloc(HP_ * 4);
  float*    scores = (float*)   alloc((size_t)B_ * T_ * 4);
  float*    ranks  = (float*)   alloc((size_t)B_ * T_ * 4);
  float*    full   = (float*)   alloc((size_t)B_ * D_ * 4);
  float*    teff   = (float*)   alloc(B_ * 4);
  float*    gst    = (float*)   alloc((size_t)R_ * B_ * T_ * 4);
  float*    pred   = (float*)   alloc((size_t)R_ * B_ * D_ * 4);
  float*    ps     = (float*)   alloc(R_ * B_ * 4);

  float* out = (float*)d_out;
  float* O_gstlast = out;                         // [B,T]
  float* O_hard    = out + B_ * T_;               // [R,B,T]
  float* O_recon   = out + B_ * T_ + R_ * B_ * T_;        // scalar
  float* O_perr    = O_recon + 1;                 // [R]
  float* O_rhoeff  = O_perr + R_;                 // [R,B]

  k_prep <<<(128 * HP_ + 255) / 256, 256, 0, stream>>>(w1, b1, w2, w1hi, w1lo, b1p, w2p);
  k_mlp  <<<(B_ * T_) / 16 / 2, 64, 0, stream>>>(emb, attn, lnw, lnb, w1hi, w1lo, b1p, w2p, b2, scores);
  k_pool <<<B_, 256, 0, stream>>>(emb, attn, full, teff);
  k_rank <<<B_, 1024, 0, stream>>>(scores, attn, ranks);
  k_gate <<<B_, 1024, 0, stream>>>(ranks, attn, teff, gst, O_hard, O_gstlast, O_rhoeff);
  k_pred <<<B_, 256, 0, stream>>>(ids, attn, gst, table, pred);
  k_cos  <<<R_ * B_, 256, 0, stream>>>(pred, full, ps);
  k_final<<<1, R_ * B_, 0, stream>>>(ps, O_recon, O_perr);
}
